// Based_40630390620259
// MI455X (gfx1250) — matrix operand order repaired and checked by reading, not compiled, not run on hardware
//
#include <hip/hip_runtime.h>

#define S_LEN 2048
#define DMODEL 1024
#define NHEAD 16
#define FDIM 16
#define HDIM 64
#define WIN 256

typedef __attribute__((ext_vector_type(4)))  float    v4f;
typedef __attribute__((ext_vector_type(8)))  float    v8f;
typedef __attribute__((ext_vector_type(8)))  _Float16 v8h;
typedef __attribute__((ext_vector_type(16))) _Float16 v16h;

static __device__ inline v8f v8f_zero() {
  v8f z;
#pragma unroll
  for (int i = 0; i < 8; ++i) z[i] = 0.0f;
  return z;
}

static __device__ inline v16h ones16h() {
  v16h r;
#pragma unroll
  for (int i = 0; i < 16; ++i) r[i] = (_Float16)1.0f;
  return r;
}

static __device__ inline v16h ld16h(const _Float16* p) {
  v8h lo = *(const v8h*)p;
  v8h hi = *(const v8h*)(p + 8);
  v16h r;
#pragma unroll
  for (int i = 0; i < 8; ++i) { r[i] = lo[i]; r[i + 8] = hi[i]; }
  return r;
}

// A fragment 16x32 f16 (ISA 7.12.2): lane m=L&15, kb=(L<16)?0:8; e<8 -> k=kb+e; e>=8 -> k=kb+8+e
static __device__ inline v16h frag_a_16x32(const _Float16* base, int ld, int lane) {
  const int m = lane & 15;
  const int kb = (lane < 16) ? 0 : 8;
  const _Float16* p = base + (size_t)m * ld + kb;
  v8h lo = *(const v8h*)p;
  v8h hi = *(const v8h*)(p + 16);
  v16h r;
#pragma unroll
  for (int i = 0; i < 8; ++i) { r[i] = lo[i]; r[i + 8] = hi[i]; }
  return r;
}

// A fragment with K=16 real features zero-padded to 32 (k>=16 are zero)
static __device__ inline v16h frag_a_16x32_pad16(const _Float16* base, int ld, int lane) {
  const int m = lane & 15;
  const int kb = (lane < 16) ? 0 : 8;
  v8h lo = *(const v8h*)(base + (size_t)m * ld + kb);
  v16h r;
#pragma unroll
  for (int i = 0; i < 8; ++i) { r[i] = lo[i]; r[i + 8] = (_Float16)0.0f; }
  return r;
}

// B fragment 32x16 f16: lane col n=L&15, kb=(L<16)?0:8; e<8 -> k=kb+e; e>=8 -> k=kb+8+e.
// Source indexed [n][k] with k contiguous (row stride ld halves).
static __device__ inline v16h frag_b_32x16(const _Float16* base, int ld, int lane) {
  return frag_a_16x32(base, ld, lane);
}

// B fragment where rows k>=16 are zero (padded 16-dim inner product)
static __device__ inline v16h frag_b_32x16_pad16(const _Float16* base, int ld, int lane) {
  return frag_a_16x32_pad16(base, ld, lane);
}

static __device__ inline v8f wmma16(v16h a, v16h b, v8f c) {
  return __builtin_amdgcn_wmma_f32_16x16x32_f16(false, a, false, b, (short)0, c, false, false);
}

// ---------------------------------------------------------------------------
// Generic GEMM: C[M,N] = A[M,K] @ B[K,N], f32 inputs, f16 WMMA compute.
// OutT=f16 (optionally transposed output C[N,M]) or OutT=float (optional +Cadd).
// Block tile 256x128x32, 256 threads = 8 waves (4x2), wave tile 64x64 (16 WMMA).
// ---------------------------------------------------------------------------
template <typename OutT, bool TRANS_OUT, bool ADD>
__global__ __launch_bounds__(256)
void gemm_wmma_k(const float* __restrict__ A, const float* __restrict__ B,
                 OutT* __restrict__ C, const float* __restrict__ Cadd,
                 int M, int N, int K) {
  __shared__ __align__(16) _Float16 As[256][40];
  __shared__ __align__(16) _Float16 Bs[128][40];  // stored transposed: [n][k]

  const int tid  = threadIdx.x;
  const int lane = tid & 31;
  const int wv   = tid >> 5;
  const int mw   = wv >> 1;   // 0..3
  const int nw   = wv & 1;    // 0..1
  const int m0   = blockIdx.y * 256;
  const int n0   = blockIdx.x * 128;

  v8f acc[4][4];
#pragma unroll
  for (int i = 0; i < 4; ++i)
#pragma unroll
    for (int j = 0; j < 4; ++j) acc[i][j] = v8f_zero();

  for (int k0 = 0; k0 < K; k0 += 32) {
    // prefetch next k-tile (global_prefetch_b8)
    if (k0 + 32 < K) {
      __builtin_prefetch(A + (size_t)(m0 + tid) * K + k0 + 32, 0, 1);
      __builtin_prefetch(B + (size_t)(k0 + 32 + (tid & 31)) * N + n0 + ((tid >> 5) << 4), 0, 1);
    }
    // stage A tile (256x32) as f16
#pragma unroll
    for (int it = 0; it < 8; ++it) {
      int idx = tid + it * 256;          // 0..2047 vec4s
      int r = idx >> 3;
      int c = (idx & 7) << 2;
      v4f a = *(const v4f*)(A + (size_t)(m0 + r) * K + k0 + c);
#pragma unroll
      for (int j = 0; j < 4; ++j) As[r][c + j] = (_Float16)a[j];
    }
    // stage B tile (32x128) transposed -> Bs[n][k]
#pragma unroll
    for (int it = 0; it < 4; ++it) {
      int idx = tid + it * 256;
      int kk = idx >> 5;
      int c  = (idx & 31) << 2;
      v4f b = *(const v4f*)(B + (size_t)(k0 + kk) * N + n0 + c);
#pragma unroll
      for (int j = 0; j < 4; ++j) Bs[c + j][kk] = (_Float16)b[j];
    }
    __syncthreads();

    v16h af[4];
#pragma unroll
    for (int i = 0; i < 4; ++i)
      af[i] = frag_a_16x32(&As[mw * 64 + i * 16][0], 40, lane);
    v16h bf[4];
#pragma unroll
    for (int j = 0; j < 4; ++j)
      bf[j] = frag_b_32x16(&Bs[nw * 64 + j * 16][0], 40, lane);
#pragma unroll
    for (int i = 0; i < 4; ++i)
#pragma unroll
      for (int j = 0; j < 4; ++j)
        acc[i][j] = wmma16(af[i], bf[j], acc[i][j]);

    __syncthreads();
  }

  const int n_lane = lane & 15;
  const int m_off  = (lane < 16) ? 0 : 8;
#pragma unroll
  for (int i = 0; i < 4; ++i)
#pragma unroll
    for (int j = 0; j < 4; ++j)
#pragma unroll
      for (int r = 0; r < 8; ++r) {
        int gm = m0 + mw * 64 + i * 16 + m_off + r;
        int gn = n0 + nw * 64 + j * 16 + n_lane;
        float v = acc[i][j][r];
        if constexpr (ADD) v += Cadd[(size_t)gm * N + gn];
        if constexpr (TRANS_OUT) C[(size_t)gn * M + gm] = (OutT)v;
        else                     C[(size_t)gm * N + gn] = (OutT)v;
      }
}

// ---------------------------------------------------------------------------
// Linear (Taylor feature map) attention, collapsed to polynomial of q.k:
//   P(a) = 1 + a/4 + a^2/32 ; out = tril(P) @ V / (rowsum(tril(P)) + eps)
// One wave per 16 queries per head. q,k: f16 [S, H*FD]; vt: f16 [D, S].
// Denominator = tril(P) @ ones, computed with an extra WMMA (no shuffles).
// ---------------------------------------------------------------------------
__global__ __launch_bounds__(128)
void lin_attn_k(const _Float16* __restrict__ qf, const _Float16* __restrict__ kf,
                const _Float16* __restrict__ vt, float* __restrict__ ctx) {
  __shared__ __align__(16) _Float16 sbuf[4][16][40];
  const int lane = threadIdx.x & 31;
  const int wv   = threadIdx.x >> 5;
  const int h    = blockIdx.y;
  const int q0   = (blockIdx.x * 4 + wv) * 16;
  _Float16 (*sb)[40] = sbuf[wv];

  const int n_lane = lane & 15;
  const int m_off  = (lane < 16) ? 0 : 8;

  v16h aq = frag_a_16x32_pad16(qf + (size_t)q0 * (NHEAD * FDIM) + h * FDIM,
                               NHEAD * FDIM, lane);
  const v16h vone = ones16h();
  v8f acc[4];
#pragma unroll
  for (int nd = 0; nd < 4; ++nd) acc[nd] = v8f_zero();
  v8f accd = v8f_zero();

  for (int kt = 0; kt <= q0 + 15; kt += 32) {
    v8f sc[2];
#pragma unroll
    for (int kj = 0; kj < 2; ++kj) {
      v16h bk = frag_b_32x16_pad16(
          kf + (size_t)(kt + kj * 16) * (NHEAD * FDIM) + h * FDIM,
          NHEAD * FDIM, lane);
      sc[kj] = wmma16(aq, bk, v8f_zero());
    }
#pragma unroll
    for (int kj = 0; kj < 2; ++kj) {
      const int key = kt + kj * 16 + n_lane;
#pragma unroll
      for (int r = 0; r < 8; ++r) {
        float a = sc[kj][r];
        float p = 1.0f + 0.25f * a + (1.0f / 32.0f) * a * a;
        if (key > q0 + m_off + r) p = 0.0f;  // causal
        sb[m_off + r][kj * 16 + n_lane] = (_Float16)p;
      }
    }
    v16h pf = frag_a_16x32(&sb[0][0], 40, lane);
    accd = wmma16(pf, vone, accd);  // denominator row-sums
#pragma unroll
    for (int nd = 0; nd < 4; ++nd) {
      v16h bv = frag_b_32x16(vt + (size_t)(h * HDIM + nd * 16) * S_LEN + kt,
                             S_LEN, lane);
      acc[nd] = wmma16(pf, bv, acc[nd]);
    }
  }

#pragma unroll
  for (int nd = 0; nd < 4; ++nd)
#pragma unroll
    for (int r = 0; r < 8; ++r)
      ctx[(size_t)(q0 + m_off + r) * DMODEL + h * HDIM + nd * 16 + n_lane] =
          acc[nd][r] / (accd[r] + 1e-9f);
}

// ---------------------------------------------------------------------------
// Banded softmax attention (j <= i + W - 1), flash-style online softmax.
// One wave per 16 queries per head. qw,kw: f16 [S,D]; vwt: f16 [D,S].
// Softmax normalizer accumulated via P @ ones WMMA; only the row-max uses
// lane butterflies.
// ---------------------------------------------------------------------------
__global__ __launch_bounds__(128)
void win_attn_k(const _Float16* __restrict__ qw, const _Float16* __restrict__ kw,
                const _Float16* __restrict__ vwt, const int* __restrict__ amask,
                float* __restrict__ ctx) {
  __shared__ __align__(16) _Float16 sbuf[4][16][40];
  const int lane = threadIdx.x & 31;
  const int wv   = threadIdx.x >> 5;
  const int h    = blockIdx.y;
  const int q0   = (blockIdx.x * 4 + wv) * 16;
  _Float16 (*sb)[40] = sbuf[wv];

  const int n_lane = lane & 15;
  const int m_off  = (lane < 16) ? 0 : 8;

  v16h aq0 = frag_a_16x32(qw + (size_t)q0 * DMODEL + h * HDIM,      DMODEL, lane);
  v16h aq1 = frag_a_16x32(qw + (size_t)q0 * DMODEL + h * HDIM + 32, DMODEL, lane);
  const v16h vone = ones16h();

  v8f acc[4];
#pragma unroll
  for (int nd = 0; nd < 4; ++nd) acc[nd] = v8f_zero();
  v8f accd = v8f_zero();
  float rmax[8];
#pragma unroll
  for (int r = 0; r < 8; ++r) rmax[r] = -1.0e30f;

  int kend = q0 + 16 + WIN - 1;
  if (kend > S_LEN) kend = S_LEN;

  for (int kt = 0; kt < kend; kt += 32) {
    v8f sc[2];
#pragma unroll
    for (int kj = 0; kj < 2; ++kj) {
      v16h bk0 = frag_b_32x16(kw + (size_t)(kt + kj * 16) * DMODEL + h * HDIM,
                              DMODEL, lane);
      v16h bk1 = frag_b_32x16(kw + (size_t)(kt + kj * 16) * DMODEL + h * HDIM + 32,
                              DMODEL, lane);
      v8f t = wmma16(aq0, bk0, v8f_zero());
      t = wmma16(aq1, bk1, t);
      const int key = kt + kj * 16 + n_lane;
      const int ok = (amask[key] != 0);
#pragma unroll
      for (int r = 0; r < 8; ++r) {
        float s = t[r] * 0.125f;  // 1/sqrt(64)
        if (!ok || key > q0 + m_off + r + (WIN - 1)) s = -1.0e30f;
        sc[kj][r] = s;
      }
    }

    float tmax[8];
#pragma unroll
    for (int r = 0; r < 8; ++r) tmax[r] = fmaxf(sc[0][r], sc[1][r]);
#pragma unroll
    for (int x = 1; x < 16; x <<= 1)
#pragma unroll
      for (int r = 0; r < 8; ++r) tmax[r] = fmaxf(tmax[r], __shfl_xor(tmax[r], x, 32));

    float al[8];
#pragma unroll
    for (int r = 0; r < 8; ++r) {
      float nm = fmaxf(rmax[r], tmax[r]);
      al[r] = __expf(rmax[r] - nm);
      rmax[r] = nm;
    }
#pragma unroll
    for (int nd = 0; nd < 4; ++nd)
#pragma unroll
      for (int r = 0; r < 8; ++r) acc[nd][r] *= al[r];
#pragma unroll
    for (int r = 0; r < 8; ++r) accd[r] *= al[r];

#pragma unroll
    for (int kj = 0; kj < 2; ++kj)
#pragma unroll
      for (int r = 0; r < 8; ++r) {
        float p = __expf(sc[kj][r] - rmax[r]);
        sb[m_off + r][kj * 16 + n_lane] = (_Float16)p;
      }

    v16h pf = frag_a_16x32(&sb[0][0], 40, lane);
    accd = wmma16(pf, vone, accd);  // softmax normalizer row-sums
#pragma unroll
    for (int nd = 0; nd < 4; ++nd) {
      v16h bv = frag_b_32x16(vwt + (size_t)(h * HDIM + nd * 16) * S_LEN + kt,
                             S_LEN, lane);
      acc[nd] = wmma16(pf, bv, acc[nd]);
    }
  }

#pragma unroll
  for (int nd = 0; nd < 4; ++nd)
#pragma unroll
    for (int r = 0; r < 8; ++r)
      ctx[(size_t)(q0 + m_off + r) * DMODEL + h * HDIM + nd * 16 + n_lane] =
          acc[nd][r] / accd[r];
}

// ---------------------------------------------------------------------------
extern "C" void kernel_launch(void* const* d_in, const int* in_sizes, int n_in,
                              void* d_out, int out_size, void* d_ws, size_t ws_size,
                              hipStream_t stream) {
  (void)in_sizes; (void)n_in; (void)out_size; (void)ws_size;
  const float* hs     = (const float*)d_in[0];
  const int*   amask  = (const int*)  d_in[1];
  const float* lin_Wq = (const float*)d_in[2];
  const float* lin_Wk = (const float*)d_in[3];
  const float* lin_Wv = (const float*)d_in[4];
  const float* lin_Wo = (const float*)d_in[5];
  const float* win_Wq = (const float*)d_in[6];
  const float* win_Wk = (const float*)d_in[7];
  const float* win_Wv = (const float*)d_in[8];
  const float* win_Wo = (const float*)d_in[9];
  float* out = (float*)d_out;

  char* ws = (char*)d_ws;
  _Float16* qlin = (_Float16*)ws; ws += (size_t)S_LEN * NHEAD * FDIM * sizeof(_Float16);
  _Float16* klin = (_Float16*)ws; ws += (size_t)S_LEN * NHEAD * FDIM * sizeof(_Float16);
  _Float16* vlt  = (_Float16*)ws; ws += (size_t)DMODEL * S_LEN * sizeof(_Float16);  // transposed
  _Float16* qwin = (_Float16*)ws; ws += (size_t)S_LEN * DMODEL * sizeof(_Float16);
  _Float16* kwin = (_Float16*)ws; ws += (size_t)S_LEN * DMODEL * sizeof(_Float16);
  _Float16* vwt  = (_Float16*)ws; ws += (size_t)DMODEL * S_LEN * sizeof(_Float16);  // transposed
  float* lctx = (float*)ws; ws += (size_t)S_LEN * DMODEL * sizeof(float);
  float* wctx = (float*)ws; ws += (size_t)S_LEN * DMODEL * sizeof(float);

  dim3 blk(256);
  // projections (f16 outputs; V projections write transposed [D,S])
  gemm_wmma_k<_Float16, false, false><<<dim3(2, 8), blk, 0, stream>>>(
      hs, lin_Wq, qlin, nullptr, S_LEN, NHEAD * FDIM, DMODEL);
  gemm_wmma_k<_Float16, false, false><<<dim3(2, 8), blk, 0, stream>>>(
      hs, lin_Wk, klin, nullptr, S_LEN, NHEAD * FDIM, DMODEL);
  gemm_wmma_k<_Float16, true, false><<<dim3(8, 8), blk, 0, stream>>>(
      hs, lin_Wv, vlt, nullptr, S_LEN, DMODEL, DMODEL);
  gemm_wmma_k<_Float16, false, false><<<dim3(8, 8), blk, 0, stream>>>(
      hs, win_Wq, qwin, nullptr, S_LEN, DMODEL, DMODEL);
  gemm_wmma_k<_Float16, false, false><<<dim3(8, 8), blk, 0, stream>>>(
      hs, win_Wk, kwin, nullptr, S_LEN, DMODEL, DMODEL);
  gemm_wmma_k<_Float16, true, false><<<dim3(8, 8), blk, 0, stream>>>(
      hs, win_Wv, vwt, nullptr, S_LEN, DMODEL, DMODEL);

  // attention branches
  lin_attn_k<<<dim3(S_LEN / 64, NHEAD), 128, 0, stream>>>(qlin, klin, vlt, lctx);
  win_attn_k<<<dim3(S_LEN / 64, NHEAD), 128, 0, stream>>>(qwin, kwin, vwt, amask, wctx);

  // output projections: out = lctx @ lin_Wo + wctx @ win_Wo
  gemm_wmma_k<float, false, false><<<dim3(8, 8), blk, 0, stream>>>(
      lctx, lin_Wo, out, nullptr, S_LEN, DMODEL, DMODEL);
  gemm_wmma_k<float, false, true><<<dim3(8, 8), blk, 0, stream>>>(
      wctx, win_Wo, out, out, S_LEN, DMODEL, DMODEL);
}
